// BACENET_71579924955202
// MI455X (gfx1250) — compile-verified, matrix operand order checked by reading, not run on hardware
//
#include <hip/hip_runtime.h>
#include <hip/hip_bf16.h>

#define PI_F   3.14159265358979f
#define RCUT_F 5.0f

typedef __attribute__((ext_vector_type(16))) _Float16 v16h;
typedef __attribute__((ext_vector_type(16))) __bf16   v16bf;
typedef __attribute__((ext_vector_type(8)))  float    v8f;

// ---------------------------------------------------------------------------
// Fragment index helper (CDNA5 ISA 7.12.2, 16-bit A/B layouts, wave32)
// element e (0..15) of the 16-bit fragment -> K index, given lane half h.
// VGPR v=e/2: v<4 -> K = (h?8:0)+2v+(e&1) ; v>=4 -> K = (h?24:16)+2(v-4)+(e&1)
// ---------------------------------------------------------------------------
__device__ __forceinline__ int frag_k(int e, int h) {
  const int v = e >> 1;
  const int base = (v < 4) ? ((h ? 8 : 0) + 2 * v) : ((h ? 24 : 16) + 2 * (v - 4));
  return base + (e & 1);
}

// A fragment (16xK tile, row m per lane) from f16 LDS, zero-padded past kmax
__device__ __forceinline__ v16h load_a_h(const _Float16* t, int ldk, int kmax,
                                         int kbase, int m, int h) {
  v16h a;
#pragma unroll
  for (int e = 0; e < 16; ++e) {
    const int k = kbase + frag_k(e, h);
    a[e] = (k < kmax) ? t[m * ldk + k] : (_Float16)0.f;
  }
  return a;
}

__device__ __forceinline__ v16bf load_a_bf(const __bf16* t, int ldk,
                                           int kbase, int m, int h) {
  v16bf a;
#pragma unroll
  for (int e = 0; e < 16; ++e)
    a[e] = t[m * ldk + kbase + frag_k(e, h)];
  return a;
}

// fast activations: single v_rcp_f32 instead of IEEE division sequence
__device__ __forceinline__ float silu_f(float x) {
  return x * __builtin_amdgcn_rcpf(1.f + __expf(-x));
}
__device__ __forceinline__ float tanh_f(float x) {
  const float t = __expf(2.f * x);
  return (t - 1.f) * __builtin_amdgcn_rcpf(t + 1.f);
}

// ---------------------------------------------------------------------------
// Kernel 0: pre-swizzle aw1/aw2 into bf16 fragment-contiguous order:
// pw[tile][lane][16] with tile = kc*8+nt, so a B fragment is one 32B load.
// ---------------------------------------------------------------------------
__global__ __launch_bounds__(256) void bace_pack_kernel(
    const float* __restrict__ aw1, const float* __restrict__ aw2,
    __bf16* __restrict__ pw1, __bf16* __restrict__ pw2) {
  const int idx = blockIdx.x * 256 + threadIdx.x;
  if (idx < 5 * 8 * 32 * 16) {  // aw1: 5 K-chunks x 8 N-tiles
    const int e = idx & 15, l = (idx >> 4) & 31, t = idx >> 9;
    const int kc = t >> 3, nt = t & 7;
    const int k = 32 * kc + frag_k(e, l >> 4);
    const int n = 16 * nt + (l & 15);
    pw1[idx] = (__bf16)aw1[k * 128 + n];
  }
  const int idx2 = idx - 5 * 8 * 32 * 16;
  if (idx2 >= 0 && idx2 < 4 * 8 * 32 * 16) {  // aw2: 4 K-chunks x 8 N-tiles
    const int e = idx2 & 15, l = (idx2 >> 4) & 31, t = idx2 >> 9;
    const int kc = t >> 3, nt = t & 7;
    const int k = 32 * kc + frag_k(e, l >> 4);
    const int n = 16 * nt + (l & 15);
    pw2[idx2] = (__bf16)aw2[k * 128 + n];
  }
}

// ---------------------------------------------------------------------------
// Kernel 1: atom species embedding  emb = tanh(onehot@sw1+sb1)@sw2+sb2
// ---------------------------------------------------------------------------
__global__ __launch_bounds__(256) void bace_embed_kernel(
    const float* __restrict__ onehot,
    const float* __restrict__ sw1, const float* __restrict__ sb1,
    const float* __restrict__ sw2, const float* __restrict__ sb2,
    float* __restrict__ emb) {
  const int a = blockIdx.x * blockDim.x + threadIdx.x;
  if (a >= 4096) return;
  float oh[8];
#pragma unroll
  for (int k = 0; k < 8; ++k) oh[k] = onehot[a * 8 + k];
  float t[16];
#pragma unroll
  for (int hh = 0; hh < 16; ++hh) {
    float s = sb1[hh];
#pragma unroll
    for (int k = 0; k < 8; ++k) s += oh[k] * sw1[k * 16 + hh];
    t[hh] = tanh_f(s);
  }
#pragma unroll
  for (int e = 0; e < 4; ++e) {
    float s = sb2[e];
#pragma unroll
    for (int hh = 0; hh < 16; ++hh) s += t[hh] * sw2[hh * 4 + e];
    emb[a * 4 + e] = s;
  }
}

// ---------------------------------------------------------------------------
// Kernel 2: per-pair radial MLP (WMMA f16) + angular scatter into G / gi2
// 8 waves/block, each wave owns 32 pairs (two 16-row WMMA M-tiles)
// ---------------------------------------------------------------------------
__global__ __launch_bounds__(256) void bace_pair_kernel(
    const float* __restrict__ rij,
    const int*   __restrict__ sidx, const int* __restrict__ fidx,
    const float* __restrict__ rw1,  const float* __restrict__ rb1,
    const float* __restrict__ rw2,  const float* __restrict__ rb2,
    const float* __restrict__ emb,
    float* __restrict__ gi2, float* __restrict__ G) {
  // fragment-ordered weight staging: [tile][lane][16] f16, 32B aligned
  __shared__ __align__(32) _Float16 s_w1p[4 * 32 * 16];  // 4 KB (rw1, kc=0)
  __shared__ __align__(32) _Float16 s_w2p[4 * 32 * 16];  // 4 KB (rw2, kc*2+nt)
  __shared__ _Float16 s_bess[8][32][8];   // 4 KB
  __shared__ _Float16 s_hid [8][16][64];  // 16 KB (per-tile reuse)
  __shared__ float    s_rad [8][32][32];  // 32 KB

  const int wave = threadIdx.x >> 5;
  const int lane = threadIdx.x & 31;
  const int m  = lane & 15;
  const int hh = lane >> 4;
  const int p  = blockIdx.x * 256 + wave * 32 + lane;

  // stage weights once per block, already in fragment order
  {
    const int tid = threadIdx.x;
    if (tid < 128) {                       // rw1: 4 N-tiles x 32 lanes
      const int l = tid & 31, nt = tid >> 5;
      const int hho = l >> 4, n = 16 * nt + (l & 15);
      v16h f;
#pragma unroll
      for (int e = 0; e < 16; ++e) {
        const int k = frag_k(e, hho);
        f[e] = (k < 8) ? (_Float16)rw1[k * 64 + n] : (_Float16)0.f;
      }
      *(v16h*)&s_w1p[tid * 16] = f;
    } else {                               // rw2: (kc*2+nt) x 32 lanes
      const int tid2 = tid - 128;
      const int l = tid2 & 31, t = tid2 >> 5;
      const int kc = t >> 1, nt = t & 1;
      const int hho = l >> 4, n = 16 * nt + (l & 15);
      v16h f;
#pragma unroll
      for (int e = 0; e < 16; ++e) {
        const int k = 32 * kc + frag_k(e, hho);
        f[e] = (_Float16)rw2[k * 32 + n];
      }
      *(v16h*)&s_w2p[tid2 * 16] = f;
    }
  }

  // geometry + bessel basis (each lane = one pair)
  const float x = rij[3 * p + 0], y = rij[3 * p + 1], z = rij[3 * p + 2];
  const float d   = sqrtf(x * x + y * y + z * z);
  const float inv = __builtin_amdgcn_rcpf(d);
  const float fc  = 0.5f * (__cosf(PI_F * d / RCUT_F) + 1.f);
#pragma unroll
  for (int n = 0; n < 8; ++n)
    s_bess[wave][lane][n] =
        (_Float16)(__sinf((float)(n + 1) * PI_F * d / RCUT_F) * inv * fc);
  __syncthreads();

  for (int tile = 0; tile < 2; ++tile) {
    // GEMM1: bessel(16x8, K padded to 32) @ rw1(8x64) -> silu -> f16 LDS
    const v16h a1 = load_a_h(&s_bess[wave][tile * 16][0], 8, 8, 0, m, hh);
#pragma unroll
    for (int nt = 0; nt < 4; ++nt) {
      const v16h b = *(const v16h*)&s_w1p[(nt * 32 + lane) * 16];
      v8f acc = {};
      acc = __builtin_amdgcn_wmma_f32_16x16x32_f16(false, a1, false, b,
                                                   (short)0, acc, false, false);
      const int col = 16 * nt + m;
#pragma unroll
      for (int i = 0; i < 8; ++i)
        s_hid[wave][i + 8 * hh][col] = (_Float16)silu_f(acc[i] + rb1[col]);
    }
    __syncthreads();

    // GEMM2: hid(16x64) @ rw2(64x32) -> silu -> s_rad
    const v16h a2k0 = load_a_h(&s_hid[wave][0][0], 64, 64, 0,  m, hh);
    const v16h a2k1 = load_a_h(&s_hid[wave][0][0], 64, 64, 32, m, hh);
#pragma unroll
    for (int nt = 0; nt < 2; ++nt) {
      v8f acc = {};
      const v16h b0 = *(const v16h*)&s_w2p[((0 * 2 + nt) * 32 + lane) * 16];
      acc = __builtin_amdgcn_wmma_f32_16x16x32_f16(false, a2k0, false, b0,
                                                   (short)0, acc, false, false);
      const v16h b1 = *(const v16h*)&s_w2p[((1 * 2 + nt) * 32 + lane) * 16];
      acc = __builtin_amdgcn_wmma_f32_16x16x32_f16(false, a2k1, false, b1,
                                                   (short)0, acc, false, false);
      const int col = 16 * nt + m;
#pragma unroll
      for (int i = 0; i < 8; ++i)
        s_rad[wave][tile * 16 + i + 8 * hh][col] = silu_f(acc[i] + rb2[col]);
    }
    __syncthreads();
  }

  // --- scatter phase: each lane owns its pair ---
  float rl[32];
#pragma unroll
  for (int k = 0; k < 32; ++k) rl[k] = s_rad[wave][lane][k];

  const int   lx[20] = {0, 0,0,1, 0,0,0,1,1,2, 0,0,0,0,1,1,1,2,2,3};
  const int   ly[20] = {0, 0,1,0, 0,1,2,0,1,0, 0,1,2,3,0,1,2,0,1,0};
  const int   lz[20] = {0, 1,0,0, 2,1,0,1,0,0, 3,2,1,0,2,1,0,1,0,0};
  const int   ls[20] = {0, 1,1,1, 2,2,2,2,2,2, 3,3,3,3,3,3,3,3,3,3};

  const float ux = x * inv + 1e-12f, uy = y * inv + 1e-12f, uz = z * inv + 1e-12f;
  const float px[4] = {1.f, ux, ux * ux, ux * ux * ux};
  const float py[4] = {1.f, uy, uy * uy, uy * uy * uy};
  const float pz[4] = {1.f, uz, uz * uz, uz * uz * uz};
  float gang[20];
#pragma unroll
  for (int c = 0; c < 20; ++c) gang[c] = px[lx[c]] * py[ly[c]] * pz[lz[c]];

  const int ai = fidx[p];
  const int aj = sidx[p];
  float ev[4];
#pragma unroll
  for (int e = 0; e < 4; ++e) ev[e] = emb[4 * aj + e];

  float* gp  = G   + ((long)ai * 32) * 20;
  float* g2p = gi2 + (long)ai * 32;
#pragma unroll
  for (int e = 0; e < 4; ++e) {
#pragma unroll
    for (int r = 0; r < 8; ++r) {
      const int er = e * 8 + r;
      const float rz[4] = {ev[e] * rl[r * 4 + 0], ev[e] * rl[r * 4 + 1],
                           ev[e] * rl[r * 4 + 2], ev[e] * rl[r * 4 + 3]};
      atomicAdd(&g2p[er], rz[0]);
#pragma unroll
      for (int c = 0; c < 20; ++c)
        atomicAdd(&gp[er * 20 + c], rz[ls[c]] * gang[c]);
    }
  }
}

// ---------------------------------------------------------------------------
// Kernel 3: per-atom gi3 contraction + MLP 160->128->128->1 (WMMA bf16)
// 4 waves/block, each wave owns 16 atoms; B from packed bf16 fragments
// ---------------------------------------------------------------------------
__global__ __launch_bounds__(128) void bace_atom_kernel(
    const float* __restrict__ gi2, const float* __restrict__ G,
    const __bf16* __restrict__ pw1, const float* __restrict__ ab1,
    const __bf16* __restrict__ pw2, const float* __restrict__ ab2,
    const float* __restrict__ aw3, const float* __restrict__ ab3,
    float* __restrict__ out) {
  __shared__ __bf16 s_feat[4][16][160];  // 20 KB
  __shared__ __bf16 s_h1  [4][16][128];  // 16 KB
  __shared__ __bf16 s_h2  [4][16][128];  // 16 KB

  const int wave = threadIdx.x >> 5;
  const int lane = threadIdx.x & 31;
  const int m  = lane & 15;
  const int hh = lane >> 4;
  const int abase = blockIdx.x * 64 + wave * 16;

  __builtin_prefetch(pw1, 0, 3);
  __builtin_prefetch(pw2, 0, 3);

  const int   ls[20]  = {0, 1,1,1, 2,2,2,2,2,2, 3,3,3,3,3,3,3,3,3,3};
  const float fct[20] = {1.f, 1.f,1.f,1.f, 1.f,2.f,1.f,2.f,2.f,1.f,
                         1.f,3.f,3.f,1.f,3.f,6.f,3.f,3.f,3.f,1.f};

  // feats: [gi2 (32) | gi3 = sum_c FACT*G^2 grouped by l (4x32)]
  for (int it = lane; it < 512; it += 32) {
    const int am = it >> 5, er = it & 31;
    const long a = abase + am;
    s_feat[wave][am][er] = (__bf16)gi2[a * 32 + er];
    const float* gp = G + (a * 32 + er) * 20;
    float acc[4] = {0.f, 0.f, 0.f, 0.f};
#pragma unroll
    for (int c = 0; c < 20; ++c) {
      const float g = gp[c];
      acc[ls[c]] += fct[c] * g * g;
    }
#pragma unroll
    for (int l = 0; l < 4; ++l)
      s_feat[wave][am][32 + l * 32 + er] = (__bf16)acc[l];
  }
  __syncthreads();

  // layer 1: feats(16x160) @ aw1(160x128), tanh
  {
    v16bf a[5];
#pragma unroll
    for (int kc = 0; kc < 5; ++kc)
      a[kc] = load_a_bf(&s_feat[wave][0][0], 160, 32 * kc, m, hh);
#pragma unroll
    for (int nt = 0; nt < 8; ++nt) {
      v8f acc = {};
#pragma unroll
      for (int kc = 0; kc < 5; ++kc) {
        const v16bf b = *(const v16bf*)(pw1 + (((kc * 8 + nt) * 32 + lane) << 4));
        acc = __builtin_amdgcn_wmma_f32_16x16x32_bf16(false, a[kc], false, b,
                                                      (short)0, acc, false, false);
      }
      const int col = 16 * nt + m;
#pragma unroll
      for (int i = 0; i < 8; ++i)
        s_h1[wave][i + 8 * hh][col] = (__bf16)tanh_f(acc[i] + ab1[col]);
    }
  }
  __syncthreads();

  // layer 2: h1(16x128) @ aw2(128x128), tanh
  {
    v16bf a[4];
#pragma unroll
    for (int kc = 0; kc < 4; ++kc)
      a[kc] = load_a_bf(&s_h1[wave][0][0], 128, 32 * kc, m, hh);
#pragma unroll
    for (int nt = 0; nt < 8; ++nt) {
      v8f acc = {};
#pragma unroll
      for (int kc = 0; kc < 4; ++kc) {
        const v16bf b = *(const v16bf*)(pw2 + (((kc * 8 + nt) * 32 + lane) << 4));
        acc = __builtin_amdgcn_wmma_f32_16x16x32_bf16(false, a[kc], false, b,
                                                      (short)0, acc, false, false);
      }
      const int col = 16 * nt + m;
#pragma unroll
      for (int i = 0; i < 8; ++i)
        s_h2[wave][i + 8 * hh][col] = (__bf16)tanh_f(acc[i] + ab2[col]);
    }
  }
  __syncthreads();

  // layer 3: 128 -> 1 dot product (no WMMA after this point, divergence OK)
  if (lane < 16) {
    float s = ab3[0];
#pragma unroll
    for (int k = 0; k < 128; ++k) s += (float)s_h2[wave][m][k] * aw3[k];
    out[abase + m] = s;
  }
}

// ---------------------------------------------------------------------------
extern "C" void kernel_launch(void* const* d_in, const int* in_sizes, int n_in,
                              void* d_out, int out_size, void* d_ws, size_t ws_size,
                              hipStream_t stream) {
  const float* rij    = (const float*)d_in[0];
  const float* onehot = (const float*)d_in[1];
  const int*   fidx   = (const int*)d_in[2];
  const int*   sidx   = (const int*)d_in[3];
  // d_in[4] = nat (4096, compile-time constant here)
  const float* sw1 = (const float*)d_in[5];
  const float* sb1 = (const float*)d_in[6];
  const float* sw2 = (const float*)d_in[7];
  const float* sb2 = (const float*)d_in[8];
  const float* rw1 = (const float*)d_in[9];
  const float* rb1 = (const float*)d_in[10];
  const float* rw2 = (const float*)d_in[11];
  const float* rb2 = (const float*)d_in[12];
  const float* aw1 = (const float*)d_in[13];
  const float* ab1 = (const float*)d_in[14];
  const float* aw2 = (const float*)d_in[15];
  const float* ab2 = (const float*)d_in[16];
  const float* aw3 = (const float*)d_in[17];
  const float* ab3 = (const float*)d_in[18];
  float* out = (float*)d_out;

  float* ws  = (float*)d_ws;
  float*  emb = ws;                          // 4096*4
  float*  gi2 = emb + 4096 * 4;              // 4096*32
  float*  G   = gi2 + 4096 * 32;             // 4096*32*20
  __bf16* pw1 = (__bf16*)(G + 4096 * 32 * 20);  // 20480 bf16 (32B aligned)
  __bf16* pw2 = pw1 + 5 * 8 * 32 * 16;          // 16384 bf16

  // zero the accumulators (graph-capture-safe)
  hipMemsetAsync(gi2, 0, (size_t)(4096 * 32 + 4096 * 32 * 20) * sizeof(float),
                 stream);

  bace_pack_kernel<<<144, 256, 0, stream>>>(aw1, aw2, pw1, pw2);
  bace_embed_kernel<<<16, 256, 0, stream>>>(onehot, sw1, sb1, sw2, sb2, emb);
  bace_pair_kernel<<<512, 256, 0, stream>>>(rij, sidx, fidx, rw1, rb1, rw2, rb2,
                                            emb, gi2, G);
  bace_atom_kernel<<<64, 128, 0, stream>>>(gi2, G, pw1, ab1, pw2, ab2, aw3, ab3,
                                           out);
}
